// TwoTowerFirstStagePolicy_24670292149142
// MI455X (gfx1250) — compile-verified, matrix operand order checked by reading, not run on hardware
//
#include <hip/hip_runtime.h>

// ---------------- problem constants ----------------
constexpr int Bn     = 1024;     // batch
constexpr int DIMC   = 512;      // context dim
constexpr int EMB    = 256;      // embedding dim
constexpr int NITEMS = 500000;   // items
constexpr int TOPK   = 100;

// ---------------- tiling constants -----------------
constexpr int NPART  = 32;               // item partitions (stage-1 grid.y)
constexpr int IPP    = NITEMS / NPART;   // 15625 items per partition
constexpr int RPW    = 32;               // rows per workgroup (2 WMMA row tiles)
constexpr int CHUNK  = 64;               // items per chunk (4 waves x 16)
constexpr int KC     = EMB / 32;         // 8 K-chunks of 32

typedef __bf16 bf16_t;
typedef bf16_t v16bf __attribute__((ext_vector_type(16)));
typedef float  v8f   __attribute__((ext_vector_type(8)));

#define NEG_INF (-__builtin_inff())

// Load the 4 float4 (16 fp32) backing one bf16 fragment for a K=32 chunk.
// Lane half h covers K = h*8..h*8+7 and 16+h*8..+7 (CDNA5 ISA §7.12.2).
__device__ inline void load_b_raw(const float* base, int h,
                                  float4& f0, float4& f1, float4& f2, float4& f3) {
    const float4* p0 = (const float4*)(base + h * 8);
    const float4* p1 = (const float4*)(base + 16 + h * 8);
    f0 = p0[0]; f1 = p0[1]; f2 = p1[0]; f3 = p1[1];
}

__device__ inline v16bf cvt_frag(float4 f0, float4 f1, float4 f2, float4 f3) {
    v16bf v;
    v[0]  = (bf16_t)f0.x; v[1]  = (bf16_t)f0.y; v[2]  = (bf16_t)f0.z; v[3]  = (bf16_t)f0.w;
    v[4]  = (bf16_t)f1.x; v[5]  = (bf16_t)f1.y; v[6]  = (bf16_t)f1.z; v[7]  = (bf16_t)f1.w;
    v[8]  = (bf16_t)f2.x; v[9]  = (bf16_t)f2.y; v[10] = (bf16_t)f2.z; v[11] = (bf16_t)f2.w;
    v[12] = (bf16_t)f3.x; v[13] = (bf16_t)f3.y; v[14] = (bf16_t)f3.z; v[15] = (bf16_t)f3.w;
    return v;
}

// ---------------- context tower (small, fp32 FMA) ----------------
__global__ void tower1_kernel(const float* __restrict__ x, const float* __restrict__ W1,
                              const float* __restrict__ b1, float* __restrict__ hbuf) {
    int idx = blockIdx.x * blockDim.x + threadIdx.x;       // over Bn*EMB
    int row = idx / EMB, col = idx % EMB;
    float acc = b1[col];
    const float* xr = x + (size_t)row * DIMC;
    #pragma unroll 4
    for (int k = 0; k < DIMC; ++k) acc = fmaf(xr[k], W1[(size_t)k * EMB + col], acc);
    hbuf[idx] = acc > 0.0f ? acc : 0.0f;
}

__global__ void tower2_kernel(const float* __restrict__ hbuf, const float* __restrict__ W2,
                              const float* __restrict__ b2, float* __restrict__ ctx) {
    int idx = blockIdx.x * blockDim.x + threadIdx.x;       // over Bn*EMB
    int row = idx / EMB, col = idx % EMB;
    float acc = b2[col];
    const float* hr = hbuf + (size_t)row * EMB;
    #pragma unroll 4
    for (int k = 0; k < EMB; ++k) acc = fmaf(hr[k], W2[(size_t)k * EMB + col], acc);
    ctx[idx] = acc;
}

// ---------------- stage 1: WMMA logits + streaming per-partition top-100 ----------------
// grid: (Bn/RPW, NPART), block: 128 threads (4 wave32)
__global__ void __launch_bounds__(128)
stage1_kernel(const float* __restrict__ ctx, const float* __restrict__ item_emb,
              float* __restrict__ cand_v, int* __restrict__ cand_i) {
    // A fragments pre-converted to bf16, one 32B fragment per (tile, kc, lane): 16 KB
    __shared__ v16bf s_A[2][KC][32];
    __shared__ float s_logits[RPW][CHUNK];          // 8 KB
    __shared__ float s_vals[RPW][TOPK];             // 12.8 KB
    __shared__ int   s_idx [RPW][TOPK];             // 12.8 KB

    const int tid  = threadIdx.x;
    const int lane = tid & 31;
    const int w    = tid >> 5;              // wave id 0..3 -> item sub-tile
    const int n    = lane & 15;             // column within 16-wide tile
    const int h    = lane >> 4;             // lane half

    const int row_base   = blockIdx.x * RPW;
    const int part       = blockIdx.y;
    const int item_begin = part * IPP;
    const int item_end   = item_begin + IPP;

    // ---- stage the A fragments (context rows, bf16) into LDS once ----
    for (int e = tid; e < 2 * KC * 32 * 16; e += blockDim.x) {
        int j    = e & 15;
        int ln   = (e >> 4) & 31;
        int kc   = (e >> 9) & (KC - 1);
        int tile = e >> 12;
        int hh = ln >> 4, mm = ln & 15;
        int col = kc * 32 + ((j < 8) ? (hh * 8 + j) : (16 + hh * 8 + (j - 8)));
        int rg  = row_base + tile * 16 + mm;
        ((bf16_t*)&s_A[tile][kc][ln])[j] = (bf16_t)ctx[(size_t)rg * EMB + col];
    }
    __syncthreads();

    // per-row top-k state lives in registers of thread `tid` (tid < RPW owns row tid)
    int   cnt  = 0;
    float minv = NEG_INF;
    int   minp = 0;

    for (int cb = item_begin; cb < item_end; cb += CHUNK) {
        const int item = cb + w * 16 + n;               // this lane's B column item
        // Clamp address in-bounds; garbage logits in padded columns are never
        // scanned (lim excludes them), so no zeroing / cndmask needed.
        const float* brow = item_emb + (size_t)((item < item_end) ? item : item_begin) * EMB;

        // prefetch next chunk's row for this lane (global_prefetch_b8)
        const int nitem = item + CHUNK;
        if (nitem < item_end)
            __builtin_prefetch(item_emb + (size_t)nitem * EMB, 0, 1);

        v8f acc0 = {}; v8f acc1 = {};

        // rolling double-buffer over K chunks: load kc+1 while computing kc
        float4 r0, r1, r2, r3;
        load_b_raw(brow, h, r0, r1, r2, r3);
        #pragma unroll
        for (int kc = 0; kc < KC; ++kc) {
            float4 n0 = r0, n1 = r1, n2 = r2, n3 = r3;
            if (kc + 1 < KC) load_b_raw(brow + (kc + 1) * 32, h, n0, n1, n2, n3);
            v16bf bb = cvt_frag(r0, r1, r2, r3);
            v16bf a0 = s_A[0][kc][lane];
            v16bf a1 = s_A[1][kc][lane];
            acc0 = __builtin_amdgcn_wmma_f32_16x16x32_bf16(
                       false, a0, false, bb, (short)0, acc0, false, false);
            acc1 = __builtin_amdgcn_wmma_f32_16x16x32_bf16(
                       false, a1, false, bb, (short)0, acc1, false, false);
            r0 = n0; r1 = n1; r2 = n2; r3 = n3;
        }

        // ---- spill D tiles to LDS (D layout: VGPR r = D[h*8+r][n]) ----
        const int col = w * 16 + n;
        #pragma unroll
        for (int r = 0; r < 8; ++r) {
            s_logits[h * 8 + r][col]      = acc0[r];
            s_logits[16 + h * 8 + r][col] = acc1[r];
        }
        __syncthreads();

        // ---- streaming top-100 update: thread `tid` owns row `tid` ----
        if (tid < RPW) {
            float* tv = s_vals[tid];
            int*   ti = s_idx[tid];
            const int lim = (item_end - cb < CHUNK) ? (item_end - cb) : CHUNK;
            for (int c = 0; c < lim; ++c) {
                float v  = s_logits[tid][c];
                int   it = cb + c;
                if (cnt < TOPK) {
                    tv[cnt] = v; ti[cnt] = it;
                    if (cnt == 0 || v < minv) { minv = v; minp = cnt; }
                    ++cnt;
                } else if (v > minv) {
                    tv[minp] = v; ti[minp] = it;
                    minv = tv[0]; minp = 0;
                    #pragma unroll 4
                    for (int q = 1; q < TOPK; ++q)
                        if (tv[q] < minv) { minv = tv[q]; minp = q; }
                }
            }
        }
        __syncthreads();
    }

    // pad (cannot trigger since IPP >> TOPK, but keeps ws fully initialized)
    if (tid < RPW) {
        for (int q = cnt; q < TOPK; ++q) { s_vals[tid][q] = NEG_INF; s_idx[tid][q] = 0x7fffffff; }
    }
    __syncthreads();

    // ---- write candidates: [row][part][k] ----
    for (int t = tid; t < RPW * TOPK; t += blockDim.x) {
        int r = t / TOPK, j = t % TOPK;
        size_t o = ((size_t)(row_base + r) * NPART + part) * TOPK + j;
        cand_v[o] = s_vals[r][j];
        cand_i[o] = s_idx[r][j];
    }
}

// ---------------- stage 2: merge 32 partial top-100 lists per row ----------------
// grid: Bn blocks, 256 threads
__global__ void __launch_bounds__(256)
merge_kernel(const float* __restrict__ cand_v, const int* __restrict__ cand_i,
             int* __restrict__ out) {
    constexpr int NC = NPART * TOPK;     // 3200 candidates
    __shared__ float cv[NC];
    __shared__ int   ci[NC];
    __shared__ float rv[256];
    __shared__ int   ri[256];
    __shared__ int   rp[256];

    const int row = blockIdx.x;
    const int tid = threadIdx.x;

    for (int j = tid; j < NC; j += 256) {
        size_t o = (size_t)row * NC + j;
        cv[j] = cand_v[o];
        ci[j] = cand_i[o];
    }
    __syncthreads();

    for (int k = 0; k < TOPK; ++k) {
        float bv = NEG_INF; int bi = 0x7fffffff; int bp = -1;
        for (int j = tid; j < NC; j += 256) {
            float v = cv[j]; int it = ci[j];
            if (v > bv || (v == bv && it < bi)) { bv = v; bi = it; bp = j; }
        }
        rv[tid] = bv; ri[tid] = bi; rp[tid] = bp;
        __syncthreads();
        for (int s = 128; s > 0; s >>= 1) {
            if (tid < s) {
                if (rv[tid + s] > rv[tid] ||
                    (rv[tid + s] == rv[tid] && ri[tid + s] < ri[tid])) {
                    rv[tid] = rv[tid + s]; ri[tid] = ri[tid + s]; rp[tid] = rp[tid + s];
                }
            }
            __syncthreads();
        }
        if (tid == 0) {
            out[(size_t)row * TOPK + k] = ri[0];
            if (rp[0] >= 0) cv[rp[0]] = NEG_INF;     // remove winner
        }
        __syncthreads();
    }
}

extern "C" void kernel_launch(void* const* d_in, const int* in_sizes, int n_in,
                              void* d_out, int out_size, void* d_ws, size_t ws_size,
                              hipStream_t stream) {
    (void)in_sizes; (void)n_in; (void)out_size; (void)ws_size;
    const float* x        = (const float*)d_in[0];
    const float* W1       = (const float*)d_in[1];
    const float* b1       = (const float*)d_in[2];
    const float* W2       = (const float*)d_in[3];
    const float* b2       = (const float*)d_in[4];
    const float* item_emb = (const float*)d_in[5];
    int* out = (int*)d_out;

    char* ws = (char*)d_ws;
    const size_t hb_bytes   = (size_t)Bn * EMB * sizeof(float);            // 1 MB
    const size_t cand_elems = (size_t)Bn * NPART * TOPK;                   // 3.2 M
    float* hbuf   = (float*)(ws);
    float* ctx    = (float*)(ws + hb_bytes);
    float* cand_v = (float*)(ws + 2 * hb_bytes);
    int*   cand_i = (int*)  (ws + 2 * hb_bytes + cand_elems * sizeof(float));

    // context tower
    tower1_kernel<<<dim3((Bn * EMB) / 256), 256, 0, stream>>>(x, W1, b1, hbuf);
    tower2_kernel<<<dim3((Bn * EMB) / 256), 256, 0, stream>>>(hbuf, W2, b2, ctx);

    // stage 1: WMMA logits + streaming per-partition top-100
    stage1_kernel<<<dim3(Bn / RPW, NPART), 128, 0, stream>>>(ctx, item_emb, cand_v, cand_i);

    // stage 2: merge partitions per row, emit sorted-descending indices
    merge_kernel<<<dim3(Bn), 256, 0, stream>>>(cand_v, cand_i, out);
}
